// Attention_10771777979079
// MI455X (gfx1250) — matrix operand order repaired and checked by reading, not compiled, not run on hardware
//
#include <hip/hip_runtime.h>
#include <hip/hip_bf16.h>
#include <stdint.h>

typedef _Float16 half_t;
typedef __attribute__((ext_vector_type(16))) _Float16 v16h;
typedef __attribute__((ext_vector_type(8)))  _Float16 v8h;
typedef __attribute__((ext_vector_type(8)))  float    v8f;
typedef __attribute__((ext_vector_type(4)))  unsigned int v4u;
typedef __attribute__((ext_vector_type(8)))  int v8i;
typedef __attribute__((ext_vector_type(4)))  int v4i;

static constexpr int S    = 2048;
static constexpr int DIM  = 4096;
static constexpr int NH   = 32;
static constexpr int NKV  = 8;
static constexpr int HD   = 128;
static constexpr int DQ   = NH * HD;    // 4096
static constexpr int DKV  = NKV * HD;   // 1024
static constexpr float SCALE = 0.08838834764831845f; // 1/sqrt(128)

// ---------------------------------------------------------------------------
// A-fragment loader: 16x32 f16 A-matrix, contiguous k order (k = 16*hi + s).
// lanes 0-15 hold K={0..15}, lanes 16-31 hold K={16..31}.
// ---------------------------------------------------------------------------
__device__ __forceinline__ v16h load_a_frag(const half_t* p, int hi) {
  return *(const v16h*)(p + hi * 16);
}

// ---------------------------------------------------------------------------
// TDM descriptor issue: 2-D tile (tile_d1 rows x tile_d0 elems) of a 2-byte
// tensor with row stride stride0 (elems), global tile base -> LDS byte addr.
// D# bitfields per CDNA5 ISA. Tile lands
// contiguously in LDS (row-major, X fastest).
// ---------------------------------------------------------------------------
__device__ __forceinline__ void tdm_load_2d(uint32_t lds_addr, const half_t* gptr,
                                            uint32_t tensor_d0, uint32_t tensor_d1,
                                            uint32_t tile_d0, uint32_t tile_d1,
                                            uint32_t stride0) {
#if defined(__HIP_DEVICE_COMPILE__)
  uint64_t ga = (uint64_t)(uintptr_t)gptr;
  v4u g0;
  g0[0] = 1u;                                              // count=1, user desc
  g0[1] = lds_addr;                                        // LDS byte address
  g0[2] = (uint32_t)ga;                                    // global_addr[31:0]
  g0[3] = (uint32_t)((ga >> 32) & 0x1FFFFFFu) | (2u << 30);// addr[56:32] | type=2
  v8i g1;
  g1[0] = (int)(1u << 16);                                 // data_size=1 (2 bytes)
  g1[1] = (int)(tensor_d0 << 16);                          // tensor_dim0[15:0]
  g1[2] = (int)((tensor_d0 >> 16) | (tensor_d1 << 16));    // dim0[31:16] | dim1[15:0]
  g1[3] = (int)((tensor_d1 >> 16) | (tile_d0 << 16));      // dim1[31:16] | tile_dim0
  g1[4] = (int)(tile_d1 & 0xFFFFu);                        // tile_dim1 | tile_dim2=0
  g1[5] = (int)stride0;                                    // dim0_stride[31:0]
  g1[6] = 0;
  g1[7] = 0;
  v4i z4 = {0, 0, 0, 0};
#if __clang_major__ >= 23
  v8i z8 = {0, 0, 0, 0, 0, 0, 0, 0};
  __builtin_amdgcn_tensor_load_to_lds(g0, g1, z4, z4, z8, 0);
#else
  __builtin_amdgcn_tensor_load_to_lds(g0, g1, z4, z4, 0);
#endif
#endif
}

__device__ __forceinline__ uint32_t lds_offset(const void* p) {
  return (uint32_t)(uintptr_t)p;  // flat LDS addr: aperture || offset (low 32b)
}

// ---------------------------------------------------------------------------
// Elementwise f32 -> f16 convert
// ---------------------------------------------------------------------------
__global__ void cvt_f32_f16(const float* __restrict__ in, half_t* __restrict__ out, int n) {
  int i = blockIdx.x * blockDim.x + threadIdx.x;
  if (i < n) out[i] = (half_t)in[i];
}

// ---------------------------------------------------------------------------
// Transpose + convert: in f32 [K][N] -> out f16 [N][K]
// ---------------------------------------------------------------------------
__global__ void transpose_cvt(const float* __restrict__ in, half_t* __restrict__ out,
                              int K, int N) {
  long long i = (long long)blockIdx.x * blockDim.x + threadIdx.x;
  if (i >= (long long)K * N) return;
  int n = (int)(i / K);
  int k = (int)(i - (long long)n * K);
  out[i] = (half_t)in[(size_t)k * N + n];
}

// ---------------------------------------------------------------------------
// RoPE + convert: in f32 [S][H][HD] -> out f16 [H][S][HD] (head-major)
// ---------------------------------------------------------------------------
__global__ void rope_to_f16(const float* __restrict__ in, half_t* __restrict__ out,
                            const float* __restrict__ cosT, const float* __restrict__ sinT,
                            int H) {
  int idx = blockIdx.x * blockDim.x + threadIdx.x;  // S*H*64 threads
  int j = idx & 63;
  int h = (idx >> 6) % H;
  int s = idx / (64 * H);
  if (s >= S) return;
  size_t ib = ((size_t)s * H + h) * HD + 2 * j;
  float t0 = in[ib], t1 = in[ib + 1];
  float c = cosT[s * 64 + j], sn = sinT[s * 64 + j];
  size_t ob = ((size_t)h * S + s) * HD + 2 * j;
  out[ob]     = (half_t)(t0 * c - t1 * sn);
  out[ob + 1] = (half_t)(t0 * sn + t1 * c);
}

// ---------------------------------------------------------------------------
// V transpose + convert: in f32 [S][NKV][HD] -> out f16 [NKV][HD][S]
// ---------------------------------------------------------------------------
__global__ void v_transpose(const float* __restrict__ in, half_t* __restrict__ out) {
  int i = blockIdx.x * blockDim.x + threadIdx.x;  // NKV*HD*S
  int s  = i & (S - 1);
  int hd = (i >> 11) & (HD - 1);
  int kv = i >> 18;
  out[i] = (half_t)in[((size_t)s * NKV + kv) * HD + hd];
}

// ---------------------------------------------------------------------------
// TDM-staged WMMA GEMM. Block = 8 waves computing a 128(M) x 64(N) tile.
// B tile (64 N-rows x 64 K-halves, 8KB) double-buffered in LDS via the
// Tensor Data Mover; A streams per-wave from global.
// ---------------------------------------------------------------------------
__global__ void gemm_wmma_tdm(const half_t* __restrict__ A, const half_t* __restrict__ BT,
                              float* __restrict__ C, int M, int N, int K, int tilesN) {
  __shared__ alignas(128) half_t bsm[2][64][64];  // 16 KB
  int tid  = threadIdx.x;
  int lane = tid & 31;
  int w    = tid >> 5;          // wave id in block (0..7)
  int l15  = lane & 15;
  int hi   = lane >> 4;
  int bm   = (blockIdx.x / tilesN) * 128;
  int tn   = blockIdx.x % tilesN;

  const half_t* Ap = A  + (size_t)(bm + w * 16 + l15) * K;
  const half_t* Bg = BT + (size_t)(tn * 64) * K;
  uint32_t lds0 = lds_offset(&bsm[0][0][0]);

  v8f acc0 = {}, acc1 = {}, acc2 = {}, acc3 = {};
  int nchunks = K >> 6;  // 64-wide K chunks

  if (tid < 32) tdm_load_2d(lds0, Bg, (uint32_t)K, (uint32_t)N, 64, 64, (uint32_t)K);

  for (int c = 0; c < nchunks; ++c) {
    if (tid < 32) {
      if (c + 1 < nchunks) {
        tdm_load_2d(lds0 + (uint32_t)(((c + 1) & 1) * 8192),
                    Bg + (size_t)(c + 1) * 64, (uint32_t)K, (uint32_t)N, 64, 64, (uint32_t)K);
        __builtin_amdgcn_s_wait_tensorcnt(1);  // oldest (chunk c) complete
      } else {
        __builtin_amdgcn_s_wait_tensorcnt(0);
      }
    }
    __syncthreads();  // chunk c visible to all waves

    const half_t* Bl = &bsm[c & 1][0][0];
#pragma unroll
    for (int ks = 0; ks < 2; ++ks) {
      int k = c * 64 + ks * 32;
      v16h a = load_a_frag(Ap + k, hi);
      const half_t* br = Bl + (size_t)l15 * 64 + ks * 32 + hi * 16;
      v16h b0 = *(const v16h*)(br);
      v16h b1 = *(const v16h*)(br + 16 * 64);
      v16h b2 = *(const v16h*)(br + 32 * 64);
      v16h b3 = *(const v16h*)(br + 48 * 64);
      acc0 = __builtin_amdgcn_wmma_f32_16x16x32_f16(false, a, false, b0, (short)0, acc0, false, false);
      acc1 = __builtin_amdgcn_wmma_f32_16x16x32_f16(false, a, false, b1, (short)0, acc1, false, false);
      acc2 = __builtin_amdgcn_wmma_f32_16x16x32_f16(false, a, false, b2, (short)0, acc2, false, false);
      acc3 = __builtin_amdgcn_wmma_f32_16x16x32_f16(false, a, false, b3, (short)0, acc3, false, false);
    }
    __syncthreads();  // all reads of buf[c&1] done before TDM overwrites it
  }

  size_t cb = (size_t)(bm + w * 16 + hi * 8) * N + tn * 64 + l15;
#pragma unroll
  for (int r = 0; r < 8; ++r) {
    C[cb + (size_t)r * N]      = acc0[r];
    C[cb + (size_t)r * N + 16] = acc1[r];
    C[cb + (size_t)r * N + 32] = acc2[r];
    C[cb + (size_t)r * N + 48] = acc3[r];
  }
}

// ---------------------------------------------------------------------------
// Causal GQA flash attention with TDM-staged K/V.
// Block = 4 waves on 4 consecutive 16-query tiles of ONE head; each 32-key
// K tile (32x128) and V tile (128x32) is TDM-loaded once and shared.
// Q f16 [NH][S][HD], K f16 [NKV][S][HD], Vt f16 [NKV][HD][S] -> out f16 [S][DQ]
// ---------------------------------------------------------------------------
__global__ void attn_wmma_tdm(const half_t* __restrict__ Qh, const half_t* __restrict__ Kh,
                              const half_t* __restrict__ Vt, half_t* __restrict__ attnOut) {
  __shared__ alignas(128) half_t Ksm[2][32][128];  // 16 KB
  __shared__ alignas(128) half_t Vsm[2][128][32];  // 16 KB
  __shared__ alignas(64)  half_t Psm[4][16][32];   // 4 KB per-wave P staging
  int tid  = threadIdx.x;
  int wid  = tid >> 5;
  int lane = tid & 31;
  int l15 = lane & 15;
  int hi  = lane >> 4;
  int qblocks = S / 64;                 // 32
  int h  = blockIdx.x / qblocks;
  int qb = blockIdx.x % qblocks;
  int q0 = qb * 64 + wid * 16;
  int kvh = h >> 2;                     // / REPEATS(4)

  const half_t* Qb = Qh + ((size_t)h * S + q0 + l15) * HD;
  const half_t* Kb = Kh + (size_t)kvh * S * HD;
  const half_t* Vb = Vt + (size_t)kvh * HD * S;
  uint32_t kls = lds_offset(&Ksm[0][0][0]);
  uint32_t vls = lds_offset(&Vsm[0][0][0]);

  v16h qf[4];
#pragma unroll
  for (int kk = 0; kk < 4; ++kk) qf[kk] = load_a_frag(Qb + kk * 32, hi);

  v8f o[8] = {};
  float m[8], l[8];
#pragma unroll
  for (int r = 0; r < 8; ++r) { m[r] = -3.0e38f; l[r] = 0.0f; }

  int mychunks = (q0 + 16 + 31) >> 5;       // this wave's causal bound
  int nmax     = (qb * 64 + 64 + 31) >> 5;  // block-uniform trip count

  if (tid < 32) {
    tdm_load_2d(kls, Kb, (uint32_t)HD, (uint32_t)S, 128, 32, (uint32_t)HD);
    tdm_load_2d(vls, Vb, (uint32_t)S, (uint32_t)HD, 32, 128, (uint32_t)S);
  }

  for (int kc = 0; kc < nmax; ++kc) {
    int kb = kc << 5;
    if (tid < 32) {
      if (kc + 1 < nmax) {
        tdm_load_2d(kls + (uint32_t)(((kc + 1) & 1) * 8192),
                    Kb + (size_t)(kb + 32) * HD, (uint32_t)HD, (uint32_t)S, 128, 32, (uint32_t)HD);
        tdm_load_2d(vls + (uint32_t)(((kc + 1) & 1) * 8192),
                    Vb + (kb + 32), (uint32_t)S, (uint32_t)HD, 32, 128, (uint32_t)S);
        __builtin_amdgcn_s_wait_tensorcnt(2);  // chunk kc's pair complete
      } else {
        __builtin_amdgcn_s_wait_tensorcnt(0);
      }
    }
    __syncthreads();  // chunk kc visible to all waves

    if (kc < mychunks) {
      const half_t* Kl = &Ksm[kc & 1][0][0];
      const half_t* Vl = &Vsm[kc & 1][0][0];
      v8f sf0 = {}, sf1 = {};
      const half_t* K0 = Kl + (size_t)l15 * 128 + hi * 16;
      const half_t* K1 = K0 + 16 * 128;
#pragma unroll
      for (int kk = 0; kk < 4; ++kk) {
        v16h b0 = *(const v16h*)(K0 + kk * 32);
        sf0 = __builtin_amdgcn_wmma_f32_16x16x32_f16(false, qf[kk], false, b0, (short)0, sf0, false, false);
      }
#pragma unroll
      for (int kk = 0; kk < 4; ++kk) {
        v16h b1 = *(const v16h*)(K1 + kk * 32);
        sf1 = __builtin_amdgcn_wmma_f32_16x16x32_f16(false, qf[kk], false, b1, (short)0, sf1, false, false);
      }
      float alpha[8];
#pragma unroll
      for (int r = 0; r < 8; ++r) {
        int row = q0 + r + hi * 8;
        float s0 = sf0[r] * SCALE;
        float s1 = sf1[r] * SCALE;
        if (kb + l15 > row)      s0 = -1.0e30f;
        if (kb + 16 + l15 > row) s1 = -1.0e30f;
        float mx = fmaxf(s0, s1);
        mx = fmaxf(mx, __shfl_xor(mx, 1, 32));
        mx = fmaxf(mx, __shfl_xor(mx, 2, 32));
        mx = fmaxf(mx, __shfl_xor(mx, 4, 32));
        mx = fmaxf(mx, __shfl_xor(mx, 8, 32));
        float mn = fmaxf(m[r], mx);
        float al = __expf(m[r] - mn);
        float p0 = __expf(s0 - mn);
        float p1 = __expf(s1 - mn);
        float ps = p0 + p1;
        ps += __shfl_xor(ps, 1, 32);
        ps += __shfl_xor(ps, 2, 32);
        ps += __shfl_xor(ps, 4, 32);
        ps += __shfl_xor(ps, 8, 32);
        l[r] = l[r] * al + ps;
        m[r] = mn;
        alpha[r] = al;
        Psm[wid][r + hi * 8][l15]      = (half_t)p0;
        Psm[wid][r + hi * 8][16 + l15] = (half_t)p1;
      }
#pragma unroll
      for (int g = 0; g < 8; ++g)
#pragma unroll
        for (int r = 0; r < 8; ++r) o[g][r] *= alpha[r];
      v16h pf = load_a_frag(&Psm[wid][l15][0], hi);  // per-wave, same-wave LDS in order
      const half_t* V0 = Vl + (size_t)l15 * 32 + hi * 16;
#pragma unroll
      for (int g = 0; g < 8; ++g) {
        v16h vf = *(const v16h*)(V0 + (size_t)g * 16 * 32);
        o[g] = __builtin_amdgcn_wmma_f32_16x16x32_f16(false, pf, false, vf, (short)0, o[g], false, false);
      }
    }
    __syncthreads();  // all reads of buf[kc&1] done before TDM overwrites it
  }

  float linv[8];
#pragma unroll
  for (int r = 0; r < 8; ++r) linv[r] = 1.0f / l[r];
  size_t ob = (size_t)(q0 + hi * 8) * DQ + (size_t)h * HD + l15;
#pragma unroll
  for (int g = 0; g < 8; ++g)
#pragma unroll
    for (int r = 0; r < 8; ++r)
      attnOut[ob + (size_t)r * DQ + g * 16] = (half_t)(o[g][r] * linv[r]);
}

// ---------------------------------------------------------------------------
static void launch_gemm(const half_t* A, const half_t* BT, float* C,
                        int M, int N, int K, hipStream_t stream) {
  int tilesN = N / 64;
  int blocks = (M / 128) * tilesN;
  gemm_wmma_tdm<<<blocks, 256, 0, stream>>>(A, BT, C, M, N, K, tilesN);
}

extern "C" void kernel_launch(void* const* d_in, const int* in_sizes, int n_in,
                              void* d_out, int out_size, void* d_ws, size_t ws_size,
                              hipStream_t stream) {
  const float* x  = (const float*)d_in[0];
  const float* wq = (const float*)d_in[1];
  const float* wk = (const float*)d_in[2];
  const float* wv = (const float*)d_in[3];
  const float* wo = (const float*)d_in[4];
  const float* fc = (const float*)d_in[5];
  const float* fs = (const float*)d_in[6];
  float* out = (float*)d_out;

  char* p = (char*)d_ws;
  auto carve = [&](size_t bytes) -> char* {
    char* r = p;
    p += (bytes + 255) & ~(size_t)255;
    return r;
  };
  half_t* xh  = (half_t*)carve((size_t)S * DIM * 2);
  half_t* wqT = (half_t*)carve((size_t)DQ * DIM * 2);
  half_t* wkT = (half_t*)carve((size_t)DKV * DIM * 2);
  half_t* wvT = (half_t*)carve((size_t)DKV * DIM * 2);
  half_t* woT = (half_t*)carve((size_t)DIM * DQ * 2);
  float*  Qf  = (float*) carve((size_t)S * DQ * 4);
  float*  Kf  = (float*) carve((size_t)S * DKV * 4);
  float*  Vf  = (float*) carve((size_t)S * DKV * 4);
  half_t* Qr  = (half_t*)carve((size_t)NH * S * HD * 2);
  half_t* Kr  = (half_t*)carve((size_t)NKV * S * HD * 2);
  half_t* Vtr = (half_t*)carve((size_t)NKV * HD * S * 2);
  half_t* At  = (half_t*)carve((size_t)S * DQ * 2);

  // stage 1: conversions / transposes
  cvt_f32_f16<<<(S * DIM) / 256, 256, 0, stream>>>(x, xh, S * DIM);
  transpose_cvt<<<(DIM * DQ)  / 256, 256, 0, stream>>>(wq, wqT, DIM, DQ);
  transpose_cvt<<<(DIM * DKV) / 256, 256, 0, stream>>>(wk, wkT, DIM, DKV);
  transpose_cvt<<<(DIM * DKV) / 256, 256, 0, stream>>>(wv, wvT, DIM, DKV);
  transpose_cvt<<<(DQ * DIM)  / 256, 256, 0, stream>>>(wo, woT, DQ, DIM);

  // stage 2: QKV projections (WMMA, TDM-staged B)
  launch_gemm(xh, wqT, Qf, S, DQ,  DIM, stream);
  launch_gemm(xh, wkT, Kf, S, DKV, DIM, stream);
  launch_gemm(xh, wvT, Vf, S, DKV, DIM, stream);

  // stage 3: RoPE + layout changes
  rope_to_f16<<<(S * NH * 64)  / 256, 256, 0, stream>>>(Qf, Qr, fc, fs, NH);
  rope_to_f16<<<(S * NKV * 64) / 256, 256, 0, stream>>>(Kf, Kr, fc, fs, NKV);
  v_transpose<<<(NKV * HD * S) / 256, 256, 0, stream>>>(Vf, Vtr);

  // stage 4: causal flash attention (WMMA, TDM-staged K/V)
  attn_wmma_tdm<<<NH * (S / 64), 128, 0, stream>>>(Qr, Kr, Vtr, At);

  // stage 5: output projection (WMMA) -> fp32 d_out
  launch_gemm(At, woT, out, S, DIM, DIM, stream);
}
